// SemanticModule_13915694039673
// MI455X (gfx1250) — matrix operand order repaired and checked by reading, not compiled, not run on hardware
//
#include <hip/hip_runtime.h>

typedef __attribute__((ext_vector_type(16))) __bf16 v16bf;
typedef __attribute__((ext_vector_type(8)))  float  v8f;

// ---------------- utility kernels ----------------

__global__ void zero_f32(float* __restrict__ p, int n) {
    int i = blockIdx.x * blockDim.x + threadIdx.x;
    if (i < n) p[i] = 0.0f;
}

// pad x: N x 6  ->  N x 32 (zeros in channels 6..31)
__global__ void pad_x(const float* __restrict__ x, float* __restrict__ h0, int n) {
    int tid = blockIdx.x * blockDim.x + threadIdx.x;
    if (tid >= n * 32) return;
    int i = tid >> 5, c = tid & 31;
    h0[tid] = (c < 6) ? x[i * 6 + c] : 0.0f;
}

// Build bf16 weights PRE-SWIZZLED into the per-lane WMMA B-fragment layout:
//   fragment element j of lane  <->  B[k = (j<8 ? j + 8*(lane>>4) : 16 + 8*(lane>>4) + (j-8))][n = nt*16 + (lane&15)]
// stored contiguously as [tile][lane][j] so a lane loads its fragment as one 32B vector.
// Conv tiles per layer (6): mat0(Wrel_tp) nt0,nt1 | mat1(Wrel_it) nt0,nt1 | mat2(Wroot0+Wroot1) nt0,nt1
// Out tiles (8): Wout nt0..nt7.  Also fused biases (b[e0]+b[e1]).
__global__ void prep_weights(const float* __restrict__ W0_rel, const float* __restrict__ W0_root,
                             const float* __restrict__ b0,     const float* __restrict__ W_rel,
                             const float* __restrict__ W_root, const float* __restrict__ b,
                             const float* __restrict__ Wout,
                             __bf16* __restrict__ wconv, __bf16* __restrict__ woutb,
                             float* __restrict__ biasc) {
    int idx = blockIdx.x * blockDim.x + threadIdx.x;
    const int CONV = 5 * 6 * 512;            // 15360
    if (idx < CONV) {
        int L = idx / 3072, rem = idx % 3072;
        int mat = rem / 1024, rem2 = rem % 1024;
        int nt = rem2 / 512, li = rem2 % 512;
        int lane = li / 16, j = li % 16;
        int k = (j < 8) ? (j + 8 * (lane >> 4)) : (16 + 8 * (lane >> 4) + (j - 8));
        int n = nt * 16 + (lane & 15);
        float v = 0.0f;
        if (L == 0) {
            if (k < 6) {
                if (mat == 0)      v = W0_rel[k * 32 + n];
                else if (mat == 1) v = W0_rel[192 + k * 32 + n];
                else               v = W0_root[k * 32 + n] + W0_root[192 + k * 32 + n];
            }
        } else {
            int Lm = L - 1;
            if (mat == 0)      v = W_rel[Lm * 2048 + k * 32 + n];
            else if (mat == 1) v = W_rel[Lm * 2048 + 1024 + k * 32 + n];
            else               v = W_root[Lm * 2048 + k * 32 + n] + W_root[Lm * 2048 + 1024 + k * 32 + n];
        }
        // tile index within layer = mat*2 + nt
        wconv[(L * 6 + mat * 2 + nt) * 512 + lane * 16 + j] = (__bf16)v;
    } else if (idx < CONV + 8 * 512) {
        int i = idx - CONV;
        int nt = i / 512, li = i % 512;
        int lane = li / 16, j = li % 16;
        int k = (j < 8) ? (j + 8 * (lane >> 4)) : (16 + 8 * (lane >> 4) + (j - 8));
        int n = nt * 16 + (lane & 15);
        woutb[nt * 512 + lane * 16 + j] = (__bf16)Wout[k * 128 + n];
    } else if (idx < CONV + 8 * 512 + 160) {
        int i = idx - (CONV + 8 * 512);
        int L = i / 32, c = i % 32;
        biasc[i] = (L == 0) ? (b0[c] + b0[32 + c])
                            : (b[(L - 1) * 64 + c] + b[(L - 1) * 64 + 32 + c]);
    }
}

// degree of 'intersects' destinations
__global__ void count_deg(const long long* __restrict__ ei, float* __restrict__ cnt, int E) {
    int e = blockIdx.x * blockDim.x + threadIdx.x;
    if (e >= E) return;
    unsafeAtomicAdd(&cnt[ei[E + e]], 1.0f);
}

__global__ void invert_deg(float* __restrict__ cnt, int n) {
    int i = blockIdx.x * blockDim.x + threadIdx.x;
    if (i < n) cnt[i] = 1.0f / fmaxf(cnt[i], 1.0f);
}

// scatter-add: 8 threads per edge, each moves a float4 (one b128 gather + 4 L2 atomics).
// Feature buffers (25.6MB) are L2-resident (192MB L2) -> atomics resolve at L2.
__global__ void scatter_add4(const float* __restrict__ feat, const long long* __restrict__ ei,
                             float* __restrict__ agg, int E) {
    long long tid = (long long)blockIdx.x * blockDim.x + threadIdx.x;
    int e = (int)(tid >> 3);
    if (e >= E) return;
    int c4 = (int)(tid & 7) * 4;
    long long s = ei[e];
    long long d = ei[E + e];
    float4 v = *(const float4*)(feat + s * 32 + c4);
    float* p = agg + d * 32 + c4;
    unsafeAtomicAdd(p + 0, v.x);
    unsafeAtomicAdd(p + 1, v.y);
    unsafeAtomicAdd(p + 2, v.z);
    unsafeAtomicAdd(p + 3, v.w);
}

// ---------------- WMMA GEMM kernels ----------------
// A fragment (16-bit A, 16x32): lane m=lane&15 is row M, half=lane>>4:
//   elem j in [0,8):  K = j + 8*half ;  elem j in [8,16): K = (j-8) + 16 + 8*half
// D (f32 16x16): lane holds rows M = r + 8*half, col N = lane&15.

__device__ __forceinline__ v16bf load_a_frag(const float* __restrict__ p, int half, float scale) {
    const float4* lo = (const float4*)(p + 8 * half);
    const float4* hi = (const float4*)(p + 16 + 8 * half);
    float4 x0 = lo[0], x1 = lo[1], y0 = hi[0], y1 = hi[1];
    v16bf a;
    a[0]  = (__bf16)(x0.x * scale); a[1]  = (__bf16)(x0.y * scale);
    a[2]  = (__bf16)(x0.z * scale); a[3]  = (__bf16)(x0.w * scale);
    a[4]  = (__bf16)(x1.x * scale); a[5]  = (__bf16)(x1.y * scale);
    a[6]  = (__bf16)(x1.z * scale); a[7]  = (__bf16)(x1.w * scale);
    a[8]  = (__bf16)(y0.x * scale); a[9]  = (__bf16)(y0.y * scale);
    a[10] = (__bf16)(y0.z * scale); a[11] = (__bf16)(y0.w * scale);
    a[12] = (__bf16)(y1.x * scale); a[13] = (__bf16)(y1.y * scale);
    a[14] = (__bf16)(y1.z * scale); a[15] = (__bf16)(y1.w * scale);
    return a;
}

#define MTILES 4

// out = relu( aggA@wA + (aggS*inv)@wB + hin@wC + bias [+ hin] )
// One wave handles MTILES 16-row tiles; 6 B fragments stay resident in VGPRs.
template <bool RESIDUAL>
__global__ void __launch_bounds__(256) conv_gemm(
        const float* __restrict__ hin,  const float* __restrict__ aggA,
        const float* __restrict__ aggS, const float* __restrict__ inv_cnt,
        const __bf16* __restrict__ wL,  const float* __restrict__ bias,
        float* __restrict__ hout, int n_nodes) {
    int wv   = (blockIdx.x * blockDim.x + threadIdx.x) >> 5;
    int lane = threadIdx.x & 31;
    int m = lane & 15, half = lane >> 4;
    int ntiles = (n_nodes + 15) >> 4;

    const v16bf* wt = (const v16bf*)wL;    // 6 tiles, each 32 fragments of 16 bf16
    v16bf bA0 = wt[0 * 32 + lane], bA1 = wt[1 * 32 + lane];
    v16bf bB0 = wt[2 * 32 + lane], bB1 = wt[3 * 32 + lane];
    v16bf bC0 = wt[4 * 32 + lane], bC1 = wt[5 * 32 + lane];
    float bv0 = bias[m], bv1 = bias[16 + m];

#pragma unroll
    for (int t = 0; t < MTILES; ++t) {
        int tile = wv * MTILES + t;
        if (tile >= ntiles) return;                     // wave-uniform
        int row0 = tile << 4;
        int row  = row0 + m; if (row >= n_nodes) row = n_nodes - 1;

        v16bf aAdd  = load_a_frag(aggA + (size_t)row * 32, half, 1.0f);
        v16bf aMean = load_a_frag(aggS + (size_t)row * 32, half, inv_cnt[row]);
        v16bf aH    = load_a_frag(hin  + (size_t)row * 32, half, 1.0f);

        v8f acc0 = {bv0, bv0, bv0, bv0, bv0, bv0, bv0, bv0};
        v8f acc1 = {bv1, bv1, bv1, bv1, bv1, bv1, bv1, bv1};
        acc0 = __builtin_amdgcn_wmma_f32_16x16x32_bf16(false, aAdd,  false, bA0, (short)0, acc0, false, false);
        acc0 = __builtin_amdgcn_wmma_f32_16x16x32_bf16(false, aMean, false, bB0, (short)0, acc0, false, false);
        acc0 = __builtin_amdgcn_wmma_f32_16x16x32_bf16(false, aH,    false, bC0, (short)0, acc0, false, false);
        acc1 = __builtin_amdgcn_wmma_f32_16x16x32_bf16(false, aAdd,  false, bA1, (short)0, acc1, false, false);
        acc1 = __builtin_amdgcn_wmma_f32_16x16x32_bf16(false, aMean, false, bB1, (short)0, acc1, false, false);
        acc1 = __builtin_amdgcn_wmma_f32_16x16x32_bf16(false, aH,    false, bC1, (short)0, acc1, false, false);

        const float* ri = hin  + ((size_t)row0 + 8 * half) * 32;
        float*       ro = hout + ((size_t)row0 + 8 * half) * 32;
        if (row0 + 16 <= n_nodes) {                     // wave-uniform fast path
#pragma unroll
            for (int r = 0; r < 8; ++r) {
                float v0 = acc0[r], v1 = acc1[r];
                if (RESIDUAL) { v0 += ri[r * 32 + m]; v1 += ri[r * 32 + 16 + m]; }
                ro[r * 32 + m]      = fmaxf(v0, 0.0f);
                ro[r * 32 + 16 + m] = fmaxf(v1, 0.0f);
            }
        } else {
#pragma unroll
            for (int r = 0; r < 8; ++r) {
                int orow = row0 + r + 8 * half;
                if (orow < n_nodes) {
                    float v0 = acc0[r], v1 = acc1[r];
                    if (RESIDUAL) { v0 += ri[r * 32 + m]; v1 += ri[r * 32 + 16 + m]; }
                    ro[r * 32 + m]      = fmaxf(v0, 0.0f);
                    ro[r * 32 + 16 + m] = fmaxf(v1, 0.0f);
                }
            }
        }
    }
}

// out = h @ Wout(32x128) + bout ; 8 B fragments resident, MTILES tiles per wave.
__global__ void __launch_bounds__(256) out_gemm(
        const float* __restrict__ hin, const __bf16* __restrict__ w,
        const float* __restrict__ bout, float* __restrict__ out, int n_nodes) {
    int wv   = (blockIdx.x * blockDim.x + threadIdx.x) >> 5;
    int lane = threadIdx.x & 31;
    int m = lane & 15, half = lane >> 4;
    int ntiles = (n_nodes + 15) >> 4;

    const v16bf* wt = (const v16bf*)w;
    v16bf bm[8];
#pragma unroll
    for (int nt = 0; nt < 8; ++nt) bm[nt] = wt[nt * 32 + lane];

#pragma unroll
    for (int t = 0; t < MTILES; ++t) {
        int tile = wv * MTILES + t;
        if (tile >= ntiles) return;                     // wave-uniform
        int row0 = tile << 4;
        int row  = row0 + m; if (row >= n_nodes) row = n_nodes - 1;

        v16bf a = load_a_frag(hin + (size_t)row * 32, half, 1.0f);
        float* ro = out + ((size_t)row0 + 8 * half) * 128;
        bool full = (row0 + 16 <= n_nodes);             // wave-uniform

#pragma unroll
        for (int nt = 0; nt < 8; ++nt) {
            float bv = bout[nt * 16 + m];
            v8f acc = {bv, bv, bv, bv, bv, bv, bv, bv};
            acc = __builtin_amdgcn_wmma_f32_16x16x32_bf16(false, a, false, bm[nt], (short)0, acc, false, false);
            if (full) {
#pragma unroll
                for (int r = 0; r < 8; ++r) ro[r * 128 + nt * 16 + m] = acc[r];
            } else {
#pragma unroll
                for (int r = 0; r < 8; ++r)
                    if (row0 + r + 8 * half < n_nodes) ro[r * 128 + nt * 16 + m] = acc[r];
            }
        }
    }
}

// ---------------- host launch ----------------

extern "C" void kernel_launch(void* const* d_in, const int* in_sizes, int n_in,
                              void* d_out, int out_size, void* d_ws, size_t ws_size,
                              hipStream_t stream) {
    const float*     x      = (const float*)d_in[0];
    const long long* ei_tp  = (const long long*)d_in[1];
    const long long* ei_it  = (const long long*)d_in[2];
    const float*     W0_rel = (const float*)d_in[3];
    const float*     W0_root= (const float*)d_in[4];
    const float*     b0     = (const float*)d_in[5];
    const float*     W_rel  = (const float*)d_in[6];
    const float*     W_root = (const float*)d_in[7];
    const float*     b      = (const float*)d_in[8];
    const float*     Wout   = (const float*)d_in[9];
    const float*     bout   = (const float*)d_in[10];
    float*           out    = (float*)d_out;

    const int N   = in_sizes[0] / 6;
    const int ETP = in_sizes[1] / 2;
    const int EIT = in_sizes[2] / 2;

    // workspace layout (all sections 32B-aligned given 32-divisible sizes)
    size_t nf = (size_t)N * 32;
    float*  h0    = (float*)d_ws;
    float*  h1    = h0 + nf;
    float*  aggA  = h1 + nf;
    float*  aggS  = aggA + nf;          // aggA/aggS contiguous -> one zero pass
    float*  cnt   = aggS + nf;          // N floats; becomes 1/max(deg,1)
    __bf16* wconv = (__bf16*)(cnt + N); // 5*6*512 bf16 pre-swizzled fragments
    __bf16* woutb = wconv + 5 * 6 * 512;
    float*  biasc = (float*)(woutb + 8 * 512);

    const int BLK = 256;

    pad_x<<<(N * 32 + BLK - 1) / BLK, BLK, 0, stream>>>(x, h0, N);

    {
        int T = 5 * 6 * 512 + 8 * 512 + 160;
        prep_weights<<<(T + BLK - 1) / BLK, BLK, 0, stream>>>(
            W0_rel, W0_root, b0, W_rel, W_root, b, Wout, wconv, woutb, biasc);
    }

    zero_f32<<<(N + BLK - 1) / BLK, BLK, 0, stream>>>(cnt, N);
    count_deg<<<(EIT + BLK - 1) / BLK, BLK, 0, stream>>>(ei_it, cnt, EIT);
    invert_deg<<<(N + BLK - 1) / BLK, BLK, 0, stream>>>(cnt, N);

    int tiles = (N + 15) / 16;
    int waves = (tiles + MTILES - 1) / MTILES;
    int gemm_blocks = (waves + 7) / 8;  // 8 wave32 per 256-thread block

    float* hin = h0;
    float* hout = h1;
    for (int L = 0; L < 5; ++L) {
        zero_f32<<<(int)((2 * nf + BLK - 1) / BLK), BLK, 0, stream>>>(aggA, (int)(2 * nf));
        long long Ttp = (long long)ETP * 8;
        scatter_add4<<<(int)((Ttp + BLK - 1) / BLK), BLK, 0, stream>>>(hin, ei_tp, aggA, ETP);
        long long Tit = (long long)EIT * 8;
        scatter_add4<<<(int)((Tit + BLK - 1) / BLK), BLK, 0, stream>>>(hin, ei_it, aggS, EIT);

        const __bf16* wLp = wconv + L * (6 * 512);
        if (L == 0)
            conv_gemm<false><<<gemm_blocks, BLK, 0, stream>>>(hin, aggA, aggS, cnt, wLp, biasc + L * 32, hout, N);
        else
            conv_gemm<true><<<gemm_blocks, BLK, 0, stream>>>(hin, aggA, aggS, cnt, wLp, biasc + L * 32, hout, N);
        float* t = hin; hin = hout; hout = t;
    }

    out_gemm<<<gemm_blocks, BLK, 0, stream>>>(hin, woutb, bout, out, N);
}
